// NETSSampler_31748398252417
// MI455X (gfx1250) — matrix operand order repaired and checked by reading, not compiled, not run on hardware
//
#include <hip/hip_runtime.h>
#include <hip/hip_bf16.h>
#include <math.h>

// ---------------------------------------------------------------------------
// NETS sampler, fused per-batch-tile persistent kernel for gfx1250 (MI455X).
// Block owns 32 batch rows for all 50 Langevin steps. All GEMMs use
// V_WMMA_F32_16X16X32_BF16. Wide GEMMs are n-blocked (4 n-tiles x 2 m-tiles
// per k-step: A loaded once per k, shared across 4 B fragments / 8 WMMAs);
// narrow GEMMs are software-pipelined with double-buffered fragments.
// ---------------------------------------------------------------------------

typedef __attribute__((ext_vector_type(16))) __bf16 v16bf;
typedef __attribute__((ext_vector_type(8)))  __bf16 v8bf;
typedef __attribute__((ext_vector_type(8)))  float  v8f;

#define B_SZ   8192
#define ZD     128
#define XD     1024
#define HD     1024
#define MT     32       // batch rows per block (two 16-row m-tiles)
#define NSTEPS 50
#define KB1    1153     // Z + X + 1 (t column)
#define LDB1   1160     // KB1 padded to multiple of 8 for 16B-aligned loads
#define EPS_C  2.0f

__device__ __forceinline__ v16bf cat8(v8bf a, v8bf b) {
  return __builtin_shufflevector(a, b, 0,1,2,3,4,5,6,7,8,9,10,11,12,13,14,15);
}

__device__ __forceinline__ v8f wmma_bf16(v16bf a, v16bf b, v8f c) {
  return __builtin_amdgcn_wmma_f32_16x16x32_bf16(
      /*neg_a=*/false, a, /*neg_b=*/false, b,
      /*c_mod=*/(short)0, c, /*reuse_a=*/false, /*reuse_b=*/false);
}

// Branch-free tanh: no EXEC divergence (keeps waves WMMA-ready), 2 trans ops (exp, rcp).
__device__ __forceinline__ float fast_tanh(float x) {
  float xc = fminf(fmaxf(x, -10.0f), 10.0f);
  float e  = __expf(2.0f * xc);
  return 1.0f - 2.0f * __builtin_amdgcn_rcpf(e + 1.0f);
}

// ---------------------------------------------------------------------------
// Wide GEMM: 4 n-tiles x 2 m-tiles (8 accumulators). Per k-step: 2 A-fragment
// loads (shared by all 4 B columns) + 4 B-fragment loads + 8 WMMAs.
// A (LDS): MT x lda row-major bf16; lane m<16 -> row m, K 0..15,
// lanes 16..31 -> K 16..31. Bt (global): B transposed (N rows x ldb);
// lane n = lane&15 -> column, lo/hi lanes K 0..15 / 16..31.
// ---------------------------------------------------------------------------
template <int K>
__device__ __forceinline__ void gemm_n4m2(const __bf16* As, int lda,
                                          const __bf16* Bt, int ldb, int n0,
                                          int lane, v8f acc[8]) {
  const int m  = lane & 15;
  const int hi = lane >> 4;
  const __bf16* arow0 = As + m * lda + hi * 16;
  const __bf16* arow1 = arow0 + 16 * lda;
  const __bf16* br    = Bt + (size_t)(n0 + m) * ldb + hi * 16;
  const size_t  bstep = (size_t)16 * ldb;
  #pragma unroll 2
  for (int k0 = 0; k0 < K; k0 += 32) {
    v16bf a0 = cat8(*(const v8bf*)(arow0 + k0), *(const v8bf*)(arow0 + k0 + 8));
    v16bf a1 = cat8(*(const v8bf*)(arow1 + k0), *(const v8bf*)(arow1 + k0 + 8));
    #pragma unroll
    for (int j = 0; j < 4; ++j) {
      const __bf16* bj = br + (size_t)j * bstep + k0;
      v16bf bf_ = cat8(*(const v8bf*)(bj), *(const v8bf*)(bj + 8));
      acc[2 * j + 0] = wmma_bf16(a0, bf_, acc[2 * j + 0]);
      acc[2 * j + 1] = wmma_bf16(a1, bf_, acc[2 * j + 1]);
    }
  }
}

// ---------------------------------------------------------------------------
// Narrow GEMM (single n-tile, 2 m-tiles), software-pipelined: fragments for
// k+1 are loaded into a rotating set before the WMMAs consuming k's fragments
// issue, so waits carry slack instead of draining to zero.
// ---------------------------------------------------------------------------
template <int K>
__device__ __forceinline__ void gemm_tile2_pipe(const __bf16* As, int lda,
                                                const __bf16* Bt, int ldb, int n0,
                                                int lane, v8f& acc0, v8f& acc1) {
  const int m  = lane & 15;
  const int hi = lane >> 4;
  const __bf16* arow0 = As + m * lda + hi * 16;
  const __bf16* arow1 = arow0 + 16 * lda;
  const __bf16* brow  = Bt + (size_t)(n0 + m) * ldb + hi * 16;

  v8bf b0  = *(const v8bf*)(brow);
  v8bf b1  = *(const v8bf*)(brow + 8);
  v8bf a00 = *(const v8bf*)(arow0);
  v8bf a01 = *(const v8bf*)(arow0 + 8);
  v8bf a10 = *(const v8bf*)(arow1);
  v8bf a11 = *(const v8bf*)(arow1 + 8);
  #pragma unroll 4
  for (int k0 = 0; k0 < K; k0 += 32) {
    v8bf nb0 = b0, nb1 = b1, na00 = a00, na01 = a01, na10 = a10, na11 = a11;
    if (k0 + 32 < K) {   // compile-time per unrolled iteration
      const int k1 = k0 + 32;
      nb0  = *(const v8bf*)(brow + k1);
      nb1  = *(const v8bf*)(brow + k1 + 8);
      na00 = *(const v8bf*)(arow0 + k1);
      na01 = *(const v8bf*)(arow0 + k1 + 8);
      na10 = *(const v8bf*)(arow1 + k1);
      na11 = *(const v8bf*)(arow1 + k1 + 8);
    }
    v16bf bf_ = cat8(b0, b1);
    acc0 = wmma_bf16(cat8(a00, a01), bf_, acc0);
    acc1 = wmma_bf16(cat8(a10, a11), bf_, acc1);
    b0 = nb0; b1 = nb1; a00 = na00; a01 = na01; a10 = na10; a11 = na11;
  }
}

// ---------------------------------------------------------------------------
// Weight prep: fp32 -> bf16 copies / transposes (runs once per launch).
// ---------------------------------------------------------------------------
__global__ void cvt_transpose_kernel(const float* __restrict__ src,
                                     __bf16* __restrict__ dst,
                                     int R, int C, int ldd) {
  int i = blockIdx.x * blockDim.x + threadIdx.x;
  if (i >= C * ldd) return;
  int c = i / ldd;
  int r = i - c * ldd;
  dst[i] = (r < R) ? (__bf16)src[(size_t)r * C + c] : (__bf16)0.0f;
}

__global__ void cvt_copy_kernel(const float* __restrict__ src,
                                __bf16* __restrict__ dst, int n) {
  int i = blockIdx.x * blockDim.x + threadIdx.x;
  if (i < n) dst[i] = (__bf16)src[i];
}

// ---------------------------------------------------------------------------
// Main fused sampler kernel. Block = 256 threads = 8 wave32; block owns batch
// rows [32*bid, 32*bid+32). For N=1024 GEMMs wave w covers columns
// [128w, 128w+128) as two groups of 4 n-tiles; for N=128 GEMMs it covers
// columns [16w, 16w+16). Every wave computes both m-tiles.
// ---------------------------------------------------------------------------
__global__ void __launch_bounds__(256, 1)
nets_sampler_kernel(const float* __restrict__ x, const float* __restrict__ z0,
                    const float* __restrict__ noise, const float* __restrict__ log_var,
                    const float* __restrict__ bd1, const float* __restrict__ bd2,
                    const float* __restrict__ bb1, const float* __restrict__ bb2,
                    const float* __restrict__ Wb1f,
                    const __bf16* __restrict__ Wd1t, const __bf16* __restrict__ Wd2t,
                    const __bf16* __restrict__ Wd2b, const __bf16* __restrict__ Wd1b,
                    const __bf16* __restrict__ Wb1t, const __bf16* __restrict__ Wb2t,
                    float* __restrict__ out) {
  extern __shared__ char smem_raw[];
  __bf16* xb   = (__bf16*)smem_raw;        // 32 x 1024 bf16 (x tile)        64KB
  __bf16* hbuf = xb   + MT * XD;           // 32 x 1024 bf16 (h, then ga)    64KB
  __bf16* sbuf = hbuf + MT * HD;           // 32 x 1024 bf16 (r, then hb)    64KB
  __bf16* zb   = sbuf + MT * HD;           // 32 x 128 bf16 (z, A-frag)       8KB
  float*  zs   = (float*)(zb + MT * ZD);   // 32 x 128 f32 (z state)         16KB
  float*  gz   = zs + MT * ZD;             // 32 x 128 f32 (ga @ Wd1^T)      16KB
  float*  bdr  = gz + MT * ZD;             // 32 x 128 f32 (drift)           16KB
  float*  bd1s = bdr + MT * ZD;            // 1024 f32                        4KB
  float*  bd2s = bd1s + HD;                // 1024 f32                        4KB
  float*  bb1s = bd2s + XD;                // 1024 f32                        4KB
  float*  wtl  = bb1s + HD;                // 1024 f32 (Wb1 t-row)            4KB
  float*  bb2s = wtl + HD;                 // 128 f32                        0.5KB

  const int tid  = threadIdx.x;
  const int lane = tid & 31;
  const int wv   = tid >> 5;
  const int row0 = blockIdx.x * MT;

  const float dt      = 1.0f / NSTEPS;
  const float dcoef   = sqrtf(2.0f * EPS_C * dt);
  const float inv_sig = 1.0f / expf(log_var[0]);

  // --- init: x tile (bf16), z0 (f32 + bf16), biases + Wb1 t-row in LDS
  for (int i = tid; i < MT * XD; i += 256)
    xb[i] = (__bf16)x[(size_t)(row0 + (i >> 10)) * XD + (i & 1023)];
  for (int i = tid; i < MT * ZD; i += 256) {
    float v = z0[(size_t)(row0 + (i >> 7)) * ZD + (i & 127)];
    zs[i] = v;
    zb[i] = (__bf16)v;
  }
  for (int i = tid; i < HD; i += 256) {
    bd1s[i] = bd1[i];
    bd2s[i] = bd2[i];
    bb1s[i] = bb1[i];
    wtl[i]  = Wb1f[(size_t)(KB1 - 1) * HD + i];
  }
  if (tid < ZD) bb2s[tid] = bb2[tid];
  __syncthreads();

  const int m   = lane & 15;
  const int mrb = (lane >> 4) * 8;  // C/D fragment row base (hi lanes +8)

  for (int s = 0; s < NSTEPS; ++s) {
    const float t = s * dt;

    // ---- GEMM1: h = tanh(z @ Wd1 + bd1)              (32x128)x(128x1024)
    for (int g = 0; g < 2; ++g) {
      const int n0 = wv * 128 + g * 64;
      v8f acc[8] = {};
      gemm_n4m2<ZD>(zb, ZD, Wd1t, ZD, n0, lane, acc);
      #pragma unroll
      for (int j = 0; j < 4; ++j) {
        const int nc = n0 + j * 16 + m;
        const float bias = bd1s[nc];
        #pragma unroll
        for (int r = 0; r < 8; ++r) {
          hbuf[(mrb + r) * HD + nc]      = (__bf16)fast_tanh(acc[2 * j][r] + bias);
          hbuf[(16 + mrb + r) * HD + nc] = (__bf16)fast_tanh(acc[2 * j + 1][r] + bias);
        }
      }
    }
    __syncthreads();

    // ---- GEMM2: mu = h @ Wd2 + bd2 ; r = (x - mu)/sigma^2 -> sbuf
    for (int g = 0; g < 2; ++g) {
      const int n0 = wv * 128 + g * 64;
      v8f acc[8] = {};
      gemm_n4m2<HD>(hbuf, HD, Wd2t, HD, n0, lane, acc);
      #pragma unroll
      for (int j = 0; j < 4; ++j) {
        const int nc = n0 + j * 16 + m;
        const float bias = bd2s[nc];
        #pragma unroll
        for (int r = 0; r < 8; ++r) {
          const int i0 = (mrb + r) * XD + nc;
          const int i1 = i0 + 16 * XD;
          sbuf[i0] = (__bf16)(((float)xb[i0] - (acc[2 * j][r] + bias)) * inv_sig);
          sbuf[i1] = (__bf16)(((float)xb[i1] - (acc[2 * j + 1][r] + bias)) * inv_sig);
        }
      }
    }
    __syncthreads();

    // ---- GEMM3: gh = r @ Wd2^T ; ga = gh*(1-h^2), overwrites hbuf in place
    //      (safe: each wave reads h and writes ga at the same element; h has
    //       no other consumer after the GEMM2 barrier)
    for (int g = 0; g < 2; ++g) {
      const int n0 = wv * 128 + g * 64;
      v8f acc[8] = {};
      gemm_n4m2<XD>(sbuf, XD, Wd2b, XD, n0, lane, acc);
      #pragma unroll
      for (int j = 0; j < 4; ++j) {
        const int nc = n0 + j * 16 + m;
        #pragma unroll
        for (int r = 0; r < 8; ++r) {
          const int i0 = (mrb + r) * HD + nc;
          const int i1 = i0 + 16 * HD;
          const float h0 = (float)hbuf[i0];
          const float h1 = (float)hbuf[i1];
          hbuf[i0] = (__bf16)(acc[2 * j][r] * (1.0f - h0 * h0));
          hbuf[i1] = (__bf16)(acc[2 * j + 1][r] * (1.0f - h1 * h1));
        }
      }
    }
    __syncthreads();

    // ---- GEMM4: gz = ga @ Wd1^T                      (32x1024)x(1024x128)
    {
      const int n0 = wv * 16;
      v8f a0 = {}, a1 = {};
      gemm_tile2_pipe<HD>(hbuf, HD, Wd1b, HD, n0, lane, a0, a1);
      #pragma unroll
      for (int r = 0; r < 8; ++r) {
        gz[(mrb + r) * ZD + n0 + m]      = a0[r];
        gz[(16 + mrb + r) * ZD + n0 + m] = a1[r];
      }
    }
    // ---- GEMM5 (same phase; reads zb/xb, writes sbuf -- disjoint from G4):
    //      hb = tanh([z, x, t] @ Wb1 + bb1) -> sbuf (r dead after G3 barrier)
    for (int g = 0; g < 2; ++g) {
      const int n0 = wv * 128 + g * 64;
      v8f acc[8] = {};
      gemm_n4m2<ZD>(zb, ZD, Wb1t,      LDB1, n0, lane, acc);  // K 0..127  (z)
      gemm_n4m2<XD>(xb, XD, Wb1t + ZD, LDB1, n0, lane, acc);  // K 128..1151 (x)
      #pragma unroll
      for (int j = 0; j < 4; ++j) {
        const int nc = n0 + j * 16 + m;
        const float bias = bb1s[nc] + t * wtl[nc];
        #pragma unroll
        for (int r = 0; r < 8; ++r) {
          sbuf[(mrb + r) * HD + nc]      = (__bf16)fast_tanh(acc[2 * j][r] + bias);
          sbuf[(16 + mrb + r) * HD + nc] = (__bf16)fast_tanh(acc[2 * j + 1][r] + bias);
        }
      }
    }
    __syncthreads();

    // ---- GEMM6: b = hb @ Wb2 + bb2                   (32x1024)x(1024x128)
    {
      const int n0 = wv * 16;
      v8f a0 = {}, a1 = {};
      gemm_tile2_pipe<HD>(sbuf, HD, Wb2t, HD, n0, lane, a0, a1);
      const float bias = bb2s[n0 + m];
      #pragma unroll
      for (int r = 0; r < 8; ++r) {
        bdr[(mrb + r) * ZD + n0 + m]      = a0[r] + bias;
        bdr[(16 + mrb + r) * ZD + n0 + m] = a1[r] + bias;
      }
    }
    __syncthreads();

    // ---- Langevin update: z += (-EPS*(z - t*gz) + b)*dt + dcoef*noise
    for (int i = tid; i < MT * ZD; i += 256) {
      const int rr = i >> 7, cc = i & 127;
      const float zv   = zs[i];
      const float grad = zv - t * gz[i];   // grad_U = z + t * (-ga @ Wd1^T)
      const float nz   = noise[(((size_t)s * B_SZ) + row0 + rr) * ZD + cc];
      const float znew = zv + (-EPS_C * grad + bdr[i]) * dt + dcoef * nz;
      zs[i] = znew;
      zb[i] = (__bf16)znew;
    }
    __syncthreads();
  }

  for (int i = tid; i < MT * ZD; i += 256)
    out[(size_t)(row0 + (i >> 7)) * ZD + (i & 127)] = zs[i];
}

// ---------------------------------------------------------------------------
// Host launcher
// ---------------------------------------------------------------------------
extern "C" void kernel_launch(void* const* d_in, const int* in_sizes, int n_in,
                              void* d_out, int out_size, void* d_ws, size_t ws_size,
                              hipStream_t stream) {
  (void)in_sizes; (void)n_in; (void)out_size; (void)ws_size;

  const float* x       = (const float*)d_in[0];
  const float* z0      = (const float*)d_in[1];
  const float* noise   = (const float*)d_in[2];
  const float* log_var = (const float*)d_in[3];
  const float* Wd1     = (const float*)d_in[4];
  const float* bd1     = (const float*)d_in[5];
  const float* Wd2     = (const float*)d_in[6];
  const float* bd2     = (const float*)d_in[7];
  const float* Wb1     = (const float*)d_in[8];
  const float* bb1     = (const float*)d_in[9];
  const float* Wb2     = (const float*)d_in[10];
  const float* bb2     = (const float*)d_in[11];

  // Workspace carve: six bf16 weight views (~7.4 MB, L2-resident).
  char* ws = (char*)d_ws;
  size_t off = 0;
  auto carve = [&](size_t bytes) {
    char* p = ws + off;
    off += (bytes + 255) & ~(size_t)255;
    return p;
  };
  __bf16* Wd1t = (__bf16*)carve((size_t)HD * ZD * 2);    // (1024 x 128)  B^T for z@Wd1
  __bf16* Wd2t = (__bf16*)carve((size_t)XD * HD * 2);    // (1024 x 1024) B^T for h@Wd2
  __bf16* Wd2b = (__bf16*)carve((size_t)HD * XD * 2);    // (1024 x 1024) B^T for r@Wd2^T (= Wd2)
  __bf16* Wd1b = (__bf16*)carve((size_t)ZD * HD * 2);    // (128 x 1024)  B^T for ga@Wd1^T (= Wd1)
  __bf16* Wb1t = (__bf16*)carve((size_t)HD * LDB1 * 2);  // (1024 x 1160) B^T for concat@Wb1 (padded)
  __bf16* Wb2t = (__bf16*)carve((size_t)ZD * HD * 2);    // (128 x 1024)  B^T for hb@Wb2

  cvt_transpose_kernel<<<(HD * ZD + 255) / 256, 256, 0, stream>>>(Wd1, Wd1t, ZD, HD, ZD);
  cvt_transpose_kernel<<<(XD * HD + 255) / 256, 256, 0, stream>>>(Wd2, Wd2t, HD, XD, HD);
  cvt_copy_kernel<<<(HD * XD + 255) / 256, 256, 0, stream>>>(Wd2, Wd2b, HD * XD);
  cvt_copy_kernel<<<(ZD * HD + 255) / 256, 256, 0, stream>>>(Wd1, Wd1b, ZD * HD);
  cvt_transpose_kernel<<<(HD * LDB1 + 255) / 256, 256, 0, stream>>>(Wb1, Wb1t, KB1, HD, LDB1);
  cvt_transpose_kernel<<<(ZD * HD + 255) / 256, 256, 0, stream>>>(Wb2, Wb2t, HD, ZD, HD);

  const size_t smem = (size_t)3 * MT * 1024 * 2   // xb, hbuf, sbuf
                    + (size_t)MT * ZD * 2         // zb
                    + (size_t)3 * MT * ZD * 4     // zs, gz, bdr
                    + (size_t)4 * 1024 * 4        // bd1s, bd2s, bb1s, wtl
                    + (size_t)ZD * 4;             // bb2s   => ~265 KB
  (void)hipFuncSetAttribute((const void*)nets_sampler_kernel,
                            hipFuncAttributeMaxDynamicSharedMemorySize, (int)smem);

  nets_sampler_kernel<<<B_SZ / MT, 256, smem, stream>>>(
      x, z0, noise, log_var, bd1, bd2, bb1, bb2, Wb1,
      Wd1t, Wd2t, Wd2b, Wd1b, Wb1t, Wb2t, (float*)d_out);
}
